// Encoder_23141283791590
// MI455X (gfx1250) — 8-bit matrix operand order repaired and checked by reading, compiled, not run on hardware
//
#include <hip/hip_runtime.h>
#include <hip/hip_bf16.h>
#include <stdint.h>

typedef __attribute__((ext_vector_type(8))) int v8i;

__global__ void quant_half_u8(const float* __restrict__ x, uint8_t* __restrict__ q, int n) {
  int i = blockIdx.x * blockDim.x + threadIdx.x;
  if (i < n) q[i] = (uint8_t)(((int)x[i]) >> 1);
}

__global__ void pack_w(const float* __restrict__ W, int8_t* __restrict__ P,
                       int K, int nchunks, int total) {
  int idx = blockIdx.x * blockDim.x + threadIdx.x;
  if (idx >= total) return;
  int b    = idx & 3;
  int v    = (idx >> 2) & 7;
  int lane = (idx >> 5) & 31;
  int top  = idx >> 10;
  int chunk  = top % nchunks;
  int octile = top / nchunks;
  int k = chunk * 64 + (v & 1) * 4 + ((v >> 1) & 1) * 16 + ((v >> 2) & 1) * 32
          + ((lane & 16) ? 8 : 0) + b;
  int m = lane & 15;
  int8_t val = 0;
  if (k < K) val = (int8_t)(int)W[(size_t)(octile * 16 + m) * K + k];
  P[idx] = val;
}

template <int NT, int KHW, int KW>
__launch_bounds__(256)
__global__ void conv_wmma_iu8(
    const uint8_t* __restrict__ in, const int8_t* __restrict__ wpk,
    const float* __restrict__ bias, const float* __restrict__ mul,
    const float* __restrict__ clp_p, int clp_i,
    const float* __restrict__ scl_p, int scl_i,
    float m_add, float m_scale, float s_add, float s_scale,
    float* __restrict__ out_f32, uint8_t* __restrict__ out_u8, int mode,
    int Cin, int IH, int IW, int OC, int ohw_sh, int ow_sh,
    int stride, int pad, int Ktot, int nchunks)
{
  const int lane = threadIdx.x & 31;
  const int wave = threadIdx.x >> 5;
  const int tileBase = (blockIdx.x * 8 + wave) * (16 * NT);
  const int octile = blockIdx.y;
  const int ohw = 1 << ohw_sh;
  const int OW  = 1 << ow_sh;
  const int inStride = Cin * IH * IW;

  int iy0[NT], ix0[NT];
  const uint8_t* inb[NT];
  int obase[NT];
#pragma unroll
  for (int t = 0; t < NT; ++t) {
    int pos = tileBase + t * 16 + (lane & 15);
    int bi  = pos >> ohw_sh;
    int rem = pos & (ohw - 1);
    int oh  = rem >> ow_sh;
    int ow  = rem & (OW - 1);
    iy0[t]   = oh * stride - pad;
    ix0[t]   = ow * stride - pad;
    inb[t]   = in + bi * inStride;
    obase[t] = ((bi * OC) << ohw_sh) + rem;
  }

  const v8i vzero = {};
  v8i acc[NT];
#pragma unroll
  for (int t = 0; t < NT; ++t) acc[t] = vzero;

  for (int c = 0; c < nchunks; ++c) {
    const int4* ap = (const int4*)(wpk + ((size_t)(octile * nchunks + c) << 10)
                                       + ((size_t)lane << 5));
    __builtin_prefetch((const void*)(ap + 64), 0, 1);
    int4 a0 = ap[0], a1 = ap[1];
    v8i A;
    A[0] = a0.x; A[1] = a0.y; A[2] = a0.z; A[3] = a0.w;
    A[4] = a1.x; A[5] = a1.y; A[6] = a1.z; A[7] = a1.w;

    const int kbase = c * 64 + ((lane & 16) ? 8 : 0);
    v8i B[NT];
#pragma unroll
    for (int v = 0; v < 8; ++v) {
      const int kb = kbase + ((v & 4) ? 32 : 0) + ((v & 2) ? 16 : 0) + (v & 1) * 4;
      uint32_t wd[NT];
#pragma unroll
      for (int t = 0; t < NT; ++t) wd[t] = 0u;
#pragma unroll
      for (int bb = 0; bb < 4; ++bb) {
        int k = kb + bb;
        if (k < Ktot) {
          int ci = k / KHW;
          int r2 = k - ci * KHW;
          int kh = r2 / KW;
          int kw = r2 - kh * KW;
#pragma unroll
          for (int t = 0; t < NT; ++t) {
            int iy = iy0[t] + kh, ix = ix0[t] + kw;
            uint32_t val = 0;
            if ((unsigned)iy < (unsigned)IH && (unsigned)ix < (unsigned)IW)
              val = inb[t][(ci * IH + iy) * IW + ix];
            wd[t] |= val << (bb * 8);
          }
        }
      }
#pragma unroll
      for (int t = 0; t < NT; ++t) B[t][v] = (int)wd[t];
    }

#pragma unroll
    for (int t = 0; t < NT; ++t)
      acc[t] = __builtin_amdgcn_wmma_i32_16x16x64_iu8(
           true, A,  false, B[t], acc[t],
           false,  false);
  }

  float clp = 0.f, scl = 0.f;
  if (mode == 0) { clp = clp_p[clp_i]; scl = scl_p[scl_i]; }
  const int m0 = (lane & 16) ? 8 : 0;
#pragma unroll
  for (int t = 0; t < NT; ++t) {
#pragma unroll
    for (int r = 0; r < 8; ++r) {
      int oc  = octile * 16 + m0 + r;
      float h = (float)(acc[t][r] + (int)bias[oc]);
      int oidx = obase[t] + (oc << ohw_sh);
      if (mode == 0) {
        float v1 = floorf((h * mul[oc] + m_add) * m_scale);
        v1 = fminf(fmaxf(v1, 0.f), clp);
        float e = floorf((v1 * scl + s_add) * s_scale);
        if (out_f32) out_f32[oidx] = e;
        out_u8[oidx] = (uint8_t)(((int)e) >> 1);
      } else if (mode == 1) {
        float e4 = floorf((h * mul[oc] + m_add) * m_scale);
        out_f32[oidx] = floorf((e4 + 8.0f) * 0.0625f);
        float a = fminf(fabsf(e4), 255.0f);
        out_u8[oidx] = (uint8_t)(((int)a) >> 1);
      } else {
        out_f32[oidx] = floorf((h * mul[oc] + m_add) * m_scale);
      }
    }
  }
}

extern "C" void kernel_launch(void* const* d_in, const int* in_sizes, int n_in,
                              void* d_out, int out_size, void* d_ws, size_t ws_size,
                              hipStream_t stream) {
  (void)in_sizes; (void)n_in; (void)out_size; (void)ws_size;

  const float* x     = (const float*)d_in[0];
  const float* W1    = (const float*)d_in[1];
  const float* b1    = (const float*)d_in[2];
  const float* W2    = (const float*)d_in[3];
  const float* b2    = (const float*)d_in[4];
  const float* W3    = (const float*)d_in[5];
  const float* b3    = (const float*)d_in[6];
  const float* W4    = (const float*)d_in[7];
  const float* b4    = (const float*)d_in[8];
  const float* pW1   = (const float*)d_in[9];
  const float* pb1   = (const float*)d_in[10];
  const float* pW2   = (const float*)d_in[11];
  const float* pb2   = (const float*)d_in[12];
  const float* pW3   = (const float*)d_in[13];
  const float* pb3   = (const float*)d_in[14];
  const float* mulsE0  = (const float*)d_in[15];
  const float* mulsE1  = (const float*)d_in[16];
  const float* mulsE2  = (const float*)d_in[17];
  const float* mulsE3  = (const float*)d_in[18];
  const float* clpE    = (const float*)d_in[19];
  const float* sclE    = (const float*)d_in[20];
  const float* mulspE0 = (const float*)d_in[21];
  const float* mulspE1 = (const float*)d_in[22];
  const float* mulspE2 = (const float*)d_in[23];
  const float* clppE   = (const float*)d_in[24];
  const float* sclpE   = (const float*)d_in[25];

  float* out   = (float*)d_out;
  float* o_y   = out;
  float* o_z   = out + 786432;
  float* o_E2  = out + 819200;
  float* o_E3  = out + 9207808;
  float* o_pE1 = out + 11304960;
  float* o_pE2 = out + 11829248;

  uint8_t* ws = (uint8_t*)d_ws;
  size_t off = 0;
  auto take = [&](size_t n) { uint8_t* p = ws + off; off += n; return p; };
  uint8_t* xq   = take(3145728);
  uint8_t* E1q  = take(33554432);
  uint8_t* E2q  = take(8388608);
  uint8_t* E3q  = take(2097152);
  uint8_t* aq   = take(786432);
  uint8_t* pE1q = take(524288);
  uint8_t* pE2q = take(131072);
  int8_t* wp1  = (int8_t*)take(16384);
  int8_t* wp2  = (int8_t*)take(409600);
  int8_t* wp3  = (int8_t*)take(409600);
  int8_t* wp4  = (int8_t*)take(614400);
  int8_t* wpp1 = (int8_t*)take(221184);
  int8_t* wpp2 = (int8_t*)take(409600);
  int8_t* wpp3 = (int8_t*)take(409600);

  quant_half_u8<<<3145728 / 256, 256, 0, stream>>>(x, xq, 3145728);

  pack_w<<<16384 / 256, 256, 0, stream>>>(W1, wp1, 75, 2, 16384);
  pack_w<<<409600 / 256, 256, 0, stream>>>(W2, wp2, 3200, 50, 409600);
  pack_w<<<409600 / 256, 256, 0, stream>>>(W3, wp3, 3200, 50, 409600);
  pack_w<<<614400 / 256, 256, 0, stream>>>(W4, wp4, 3200, 50, 614400);
  pack_w<<<221184 / 256, 256, 0, stream>>>(pW1, wpp1, 1728, 27, 221184);
  pack_w<<<409600 / 256, 256, 0, stream>>>(pW2, wpp2, 3200, 50, 409600);
  pack_w<<<409600 / 256, 256, 0, stream>>>(pW3, wpp3, 3200, 50, 409600);

  conv_wmma_iu8<4, 25, 5><<<dim3(512, 8), 256, 0, stream>>>(
      xq, wp1, b1, mulsE0, clpE, 0, sclE, 0,
      524288.f, 1.f / 1048576.f, 262144.f, 1.f / 524288.f,
      nullptr, E1q, 0,
      3, 512, 512, 128,  16,  8, 2, 2, 75, 2);
  conv_wmma_iu8<4, 25, 5><<<dim3(128, 8), 256, 0, stream>>>(
      E1q, wp2, b2, mulsE1, clpE, 1, sclE, 1,
      32768.f, 1.f / 65536.f, 262144.f, 1.f / 524288.f,
      o_E2, E2q, 0,
      128, 256, 256, 128,  14,  7, 2, 2, 3200, 50);
  conv_wmma_iu8<4, 25, 5><<<dim3(32, 8), 256, 0, stream>>>(
      E2q, wp3, b3, mulsE2, clpE, 2, sclE, 2,
      32768.f, 1.f / 65536.f, 262144.f, 1.f / 524288.f,
      o_E3, E3q, 0,
      128, 128, 128, 128,  12,  6, 2, 2, 3200, 50);
  conv_wmma_iu8<4, 25, 5><<<dim3(8, 12), 256, 0, stream>>>(
      E3q, wp4, b4, mulsE3, nullptr, 0, nullptr, 0,
      16384.f, 1.f / 32768.f, 0.f, 0.f,
      o_y, aq, 1,
      128, 64, 64, 192,  10,  5, 2, 2, 3200, 50);
  conv_wmma_iu8<4, 9, 3><<<dim3(8, 8), 256, 0, stream>>>(
      aq, wpp1, pb1, mulspE0, clppE, 0, sclpE, 0,
      1024.f, 1.f / 2048.f, 1048576.f, 1.f / 2097152.f,
      o_pE1, pE1q, 0,
      192, 32, 32, 128,  10,  5, 1, 1, 1728, 27);
  conv_wmma_iu8<4, 25, 5><<<dim3(2, 8), 256, 0, stream>>>(
      pE1q, wpp2, pb2, mulspE1, clppE, 1, sclpE, 1,
      8192.f, 1.f / 16384.f, 1048576.f, 1.f / 2097152.f,
      o_pE2, pE2q, 0,
      128, 32, 32, 128,  8,  4, 2, 2, 3200, 50);
  conv_wmma_iu8<1, 25, 5><<<dim3(2, 8), 256, 0, stream>>>(
      pE2q, wpp3, pb3, mulspE2, nullptr, 0, nullptr, 0,
      4194304.f, 1.f / 8388608.f, 0.f, 0.f,
      o_z, nullptr, 2,
      128, 16, 16, 128,  6,  3, 2, 2, 3200, 50);
}
